// SpatialAttention_78958678769948
// MI455X (gfx1250) — matrix operand order repaired and checked by reading, not compiled, not run on hardware
//
#include <hip/hip_runtime.h>
#include <hip/hip_bf16.h>

typedef __attribute__((ext_vector_type(16))) __bf16 v16bf;
typedef __attribute__((ext_vector_type(8)))  __bf16 v8bf;
typedef __attribute__((ext_vector_type(4)))  __bf16 v4bf;
typedef __attribute__((ext_vector_type(8)))  float  v8f;
typedef __attribute__((ext_vector_type(4)))  float  v4f;
typedef __attribute__((ext_vector_type(4)))  unsigned v4u;
typedef __attribute__((ext_vector_type(8)))  int    v8i;
typedef __attribute__((ext_vector_type(4)))  int    v4i;

#define E_DIM   768
#define HEADS   12
#define HD_DIM  64
#define P_TOK   256
#define M_TOTAL 32768   // B*T*P = 8*16*256

__device__ __forceinline__ __bf16 f2bf(float f) {
  unsigned u = __builtin_bit_cast(unsigned, f);
  unsigned r = u + 0x7FFFu + ((u >> 16) & 1u);   // round-to-nearest-even
  unsigned short h = (unsigned short)(r >> 16);
  return __builtin_bit_cast(__bf16, h);
}

__device__ __forceinline__ v16bf cat8(v8bf lo, v8bf hi) {
  return __builtin_shufflevector(lo, hi, 0,1,2,3,4,5,6,7,8,9,10,11,12,13,14,15);
}

__device__ __forceinline__ v8f wmma_bf16(v16bf a, v16bf b, v8f c) {
  // D = A(16x32 bf16) * B(32x16 bf16) + C(16x16 f32)
  return __builtin_amdgcn_wmma_f32_16x16x32_bf16(false, a, false, b,
                                                 (short)0, c, false, false);
}

// A fragment (16x32 bf16), source row-major with leading dim ld.
// lane L holds row L%16; two contiguous 8-elem runs at kb and 8+kb, kb=(L/16)*16.
__device__ __forceinline__ v16bf load_a_frag(const __bf16* base, int ld) {
  const int lane = threadIdx.x & 31;
  const __bf16* p = base + (size_t)(lane & 15) * ld + ((lane >> 4) << 4);
  return cat8(*(const v8bf*)p, *(const v8bf*)(p + 8));
}

// B fragment (32x16 bf16) from K-major storage Bt[n][k] (ld k's per n).
// lane L holds col L%16, K-range (L/16)*16..+15 -> one contiguous 32B load.
__device__ __forceinline__ v16bf load_b_fragT(const __bf16* base, int ld) {
  const int lane = threadIdx.x & 31;
  return *(const v16bf*)(base + (size_t)(lane & 15) * ld + ((lane >> 4) << 4));
}

// ---------------------------------------------------------------------------
// Tensor Data Mover: 2-D tile load, global -> LDS. Element size = 2 bytes.
// Descriptor layout per CDNA5 ISA ch.8 (group0 128b, group1 256b, g2/g3 = 0).
// All dims/strides in data_size (element) units. Issue from ONE wave only.
// This toolchain exposes the 6-arg builtin:
//   (uint32x4 g0, int32x8 g1, int32x4 g2, int32x4 g3, int32x8, i32 cpol)
// ---------------------------------------------------------------------------
__device__ __forceinline__ void tdm_load_2d_bf16(
    unsigned lds_off, const __bf16* gaddr,
    unsigned tile_d0, unsigned tile_d1,
    unsigned tensor_d0, unsigned tensor_d1,
    unsigned stride0) {
  const unsigned long long ga = (unsigned long long)(size_t)gaddr;
  v4u g0;
  g0[0] = 1u;                                   // count=1 (valid user desc)
  g0[1] = lds_off;                              // lds_addr (bytes)
  g0[2] = (unsigned)(ga & 0xffffffffu);         // global_addr[31:0]
  g0[3] = (unsigned)((ga >> 32) & 0x01ffffffu)  // global_addr[56:32]
          | (2u << 30);                         // type = 2 ("image")
  v8i g1;
  g1[0] = (int)(1u << 16);                      // data_size=1 (2B), wg_mask=0
  g1[1] = (int)((tensor_d0 & 0xffffu) << 16);   // tensor_dim0[15:0]
  g1[2] = (int)((tensor_d0 >> 16) | ((tensor_d1 & 0xffffu) << 16));
  g1[3] = (int)((tensor_d1 >> 16) | (tile_d0 << 16));   // tile_dim0
  g1[4] = (int)(tile_d1 & 0xffffu);             // tile_dim1 (tile_dim2=0)
  g1[5] = (int)stride0;                         // tensor_dim0_stride[31:0]
  g1[6] = 0;                                    // stride0 hi / stride1 lo
  g1[7] = 0;                                    // stride1 hi (2-D: unused)
  v4i z4; z4[0] = 0; z4[1] = 0; z4[2] = 0; z4[3] = 0;
  v8i z8; z8[0] = 0; z8[1] = 0; z8[2] = 0; z8[3] = 0;
  z8[4] = 0; z8[5] = 0; z8[6] = 0; z8[7] = 0;
  __builtin_amdgcn_tensor_load_to_lds(g0, g1, z4, z4, z8, 0);
}

// Flat shared-memory pointer -> LDS byte offset (low 32 bits of the address).
__device__ __forceinline__ unsigned lds_off_of(const void* p) {
  return (unsigned)(size_t)p;
}

// ---------------------------------------------------------------------------
// f32 -> bf16 bulk convert (x once; weights are converted in-kernel)
// ---------------------------------------------------------------------------
__global__ __launch_bounds__(256) void cvt_bf16_kernel(const float* __restrict__ in,
                                                       __bf16* __restrict__ out,
                                                       int n4) {
  int i = blockIdx.x * blockDim.x + threadIdx.x;
  if (i >= n4) return;
  v4f t = *(const v4f*)(in + 4 * (size_t)i);
  v4bf o;
  o[0] = f2bf(t[0]); o[1] = f2bf(t[1]); o[2] = f2bf(t[2]); o[3] = f2bf(t[3]);
  *(v4bf*)(out + 4 * (size_t)i) = o;
}

// ---------------------------------------------------------------------------
// Tiled GEMM:  out[M=32768][768] = A[32768][768](bf16) * W[768][768](f32)
// MODE 0: +bias, 2D-RoPE, *0.125 (Q)   MODE 1: +bias, 2D-RoPE (K)
// MODE 2: +bias (V)                    MODE 3: plain, f32 output (Wo proj)
// Block: 256 threads = 8 waves; block tile 128x128; wave tile 32x64.
// Double-buffered LDS: A tile via TDM (wave 0), W tile via VALU cvt stores.
// ---------------------------------------------------------------------------
template <int MODE>
__global__ __launch_bounds__(256) void gemm_kernel(
    const __bf16* __restrict__ A,
    const float*  __restrict__ W,
    const float*  __restrict__ bias,
    __bf16* __restrict__ out_bf,
    float*  __restrict__ out_f) {
  __shared__ __align__(32) __bf16 As[2][128][32];   // row-major  [m][k]
  __shared__ __align__(32) __bf16 Bs[2][128][32];   // K-major    [n][k]
  __shared__ float ctab[256];
  __shared__ float stab[256];

  const int tid  = threadIdx.x;
  const int lane = tid & 31;
  const int wid  = tid >> 5;
  const int wrow = (wid & 3) << 5;   // 0,32,64,96
  const int wcol = (wid >> 2) << 6;  // 0,64
  const int m0 = blockIdx.x << 7;
  const int n0 = blockIdx.y << 7;

  if (MODE < 2) {  // RoPE table: theta = pos * 10000^(-2i/32), pos,i in [0,16)
    const int pos = tid >> 4, fi = tid & 15;
    const float invf = __powf(10000.f, -(float)(2 * fi) / 32.f);
    const float th = (float)pos * invf;
    ctab[tid] = __cosf(th);
    stab[tid] = __sinf(th);
  }

  v8f acc[2][4];
#pragma unroll
  for (int mt = 0; mt < 2; ++mt)
#pragma unroll
    for (int nt = 0; nt < 4; ++nt)
#pragma unroll
      for (int r = 0; r < 8; ++r) acc[mt][nt][r] = 0.f;

  const int brow = tid >> 3, bseg = (tid & 7) << 4;

  // Stage tiles for k-step k0 into buffer `buf`.
  auto stage = [&](int k0, int buf) {
    if (tid < 32) {  // wave 0 drives the Tensor Data Mover for the A tile
      tdm_load_2d_bf16(lds_off_of(&As[buf][0][0]),
                       A + (size_t)m0 * E_DIM + k0,
                       /*tile_d0=*/32, /*tile_d1=*/128,
                       /*tensor_d0=*/E_DIM - k0, /*tensor_d1=*/128,
                       /*stride0=*/E_DIM);
    }
    // W tile: 32x128 f32 -> bf16, stored K-major for contiguous B-fragments
    const float* src = W + (size_t)(k0 + brow) * E_DIM + n0 + bseg;
#pragma unroll
    for (int q = 0; q < 4; ++q) {
      v4f t = *(const v4f*)(src + 4 * q);
#pragma unroll
      for (int e = 0; e < 4; ++e) Bs[buf][bseg + 4 * q + e][brow] = f2bf(t[e]);
    }
    if (k0 + 32 < E_DIM) __builtin_prefetch(src + 32 * E_DIM, 0, 1);
  };

  stage(0, 0);
  for (int i = 0; i < E_DIM / 32; ++i) {
    const int cur = i & 1;
    if (tid < 32) __builtin_amdgcn_s_wait_tensorcnt(0);  // A(cur) DMA done
    __syncthreads();                                     // B(cur) stores done
    if (i + 1 < E_DIM / 32) stage((i + 1) << 5, cur ^ 1); // overlap next stage

    v16bf bf[4];
#pragma unroll
    for (int nt = 0; nt < 4; ++nt)
      bf[nt] = load_b_fragT(&Bs[cur][wcol + nt * 16][0], 32);
#pragma unroll
    for (int mt = 0; mt < 2; ++mt) {
      v16bf af = load_a_frag(&As[cur][wrow + mt * 16][0], 32);
#pragma unroll
      for (int nt = 0; nt < 4; ++nt)
        acc[mt][nt] = wmma_bf16(af, bf[nt], acc[mt][nt]);
    }
  }

  // Epilogue: bias (+ RoPE for Q/K). D layout: lane holds col = n&15,
  // rows r + 8*(lane/16). RoPE pair (2i,2i+1) = adjacent lanes -> shfl_xor(1).
#pragma unroll
  for (int mt = 0; mt < 2; ++mt) {
#pragma unroll
    for (int nt = 0; nt < 4; ++nt) {
      const int col   = n0 + wcol + nt * 16 + (lane & 15);
      const int rbase = m0 + wrow + mt * 16 + ((lane >> 4) << 3);
      const float bv  = (MODE == 3) ? 0.f : bias[col];
      const int dh  = col & 63;          // dim within head
      const int isy = dh >> 5;           // 0: x-rope half, 1: y-rope half
      const int fi  = (dh & 31) >> 1;    // frequency index
      const int odd = dh & 1;
#pragma unroll
      for (int r = 0; r < 8; ++r) {
        const int m = rbase + r;
        float v = acc[mt][nt][r] + bv;
        if (MODE < 2) {
          const int p   = m & 255;                    // token within 16x16 grid
          const int pos = isy ? (p >> 4) : (p & 15);  // py : px
          const float c = ctab[pos * 16 + fi];
          const float s = stab[pos * 16 + fi];
          const float partner = __shfl_xor(v, 1, 32);
          v = odd ? fmaf(v, c, partner * s) : fmaf(v, c, -partner * s);
          if (MODE == 0) v *= 0.125f;                 // 1/sqrt(64)
        }
        if (MODE == 3) out_f[(size_t)m * E_DIM + col] = v;
        else           out_bf[(size_t)m * E_DIM + col] = f2bf(v);
      }
    }
  }
}

// ---------------------------------------------------------------------------
// Flash attention, one (b,t,head) per block: 256 queries x 256 keys, HD=64.
// 8 waves x 32 query rows; stream 32-key blocks through LDS (K via TDM).
// ---------------------------------------------------------------------------
__global__ __launch_bounds__(256) void attn_kernel(
    const __bf16* __restrict__ Q, const __bf16* __restrict__ K,
    const __bf16* __restrict__ V, __bf16* __restrict__ Y) {
  __shared__ __align__(32) __bf16 Ks[32][64];       // [key][d]
  __shared__ __align__(32) __bf16 Vs[64][32];       // [d][key] (transposed)
  __shared__ __align__(32) __bf16 Ps[8][32][32];    // per-wave P staging

  const int tid = threadIdx.x, lane = tid & 31, wid = tid >> 5;
  const int bt = blockIdx.x / HEADS;
  const int h  = blockIdx.x % HEADS;
  const size_t base = (size_t)bt * P_TOK * E_DIM + (size_t)h * HD_DIM;
  const int qrow = wid << 5;

  // Resident Q fragments [mt][ks] as A-frags, contiguous k order (k = 16h + s), built straight from global.
  v16bf qf[2][2];
#pragma unroll
  for (int mt = 0; mt < 2; ++mt)
#pragma unroll
    for (int ks = 0; ks < 2; ++ks) {
      const __bf16* p = Q + base +
          (size_t)(qrow + mt * 16 + (lane & 15)) * E_DIM +
          (ks << 5) + ((lane >> 4) << 4);
      qf[mt][ks] = cat8(*(const v8bf*)p, *(const v8bf*)(p + 8));
    }

  v8f O[2][4];
  float mrun[2][8], lrun[2][8];
#pragma unroll
  for (int mt = 0; mt < 2; ++mt)
#pragma unroll
    for (int r = 0; r < 8; ++r) {
      mrun[mt][r] = -3.0e38f;
      lrun[mt][r] = 0.f;
#pragma unroll
      for (int nt = 0; nt < 4; ++nt) O[mt][nt][r] = 0.f;
    }

  const int krow = tid >> 3, kseg = (tid & 7) << 3;

  for (int kb0 = 0; kb0 < P_TOK; kb0 += 32) {
    __syncthreads();
    // K tile 32 keys x 64 d via Tensor Data Mover (wave 0)
    if (tid < 32) {
      tdm_load_2d_bf16(lds_off_of(&Ks[0][0]),
                       K + base + (size_t)kb0 * E_DIM,
                       /*tile_d0=*/HD_DIM, /*tile_d1=*/32,
                       /*tensor_d0=*/HD_DIM, /*tensor_d1=*/P_TOK - kb0,
                       /*stride0=*/E_DIM);
    }
    {  // V transposed (d-major) so PV B-fragments are contiguous
      v8bf vv = *(const v8bf*)(V + base + (size_t)(kb0 + krow) * E_DIM + kseg);
#pragma unroll
      for (int e = 0; e < 8; ++e) Vs[kseg + e][krow] = vv[e];
    }
    if (tid < 32) __builtin_amdgcn_s_wait_tensorcnt(0);
    __syncthreads();

    v16bf kf[2][2];  // [key-tile][d-chunk], B-frag: N=key, K-dim=d
#pragma unroll
    for (int kt = 0; kt < 2; ++kt)
#pragma unroll
      for (int ks = 0; ks < 2; ++ks)
        kf[kt][ks] = *(const v16bf*)&Ks[kt * 16 + (lane & 15)]
                                      [(ks << 5) + ((lane >> 4) << 4)];
    v16bf vfr[4];    // B-frag: N=d, K-dim=key
#pragma unroll
    for (int nt = 0; nt < 4; ++nt)
      vfr[nt] = *(const v16bf*)&Vs[nt * 16 + (lane & 15)][(lane >> 4) << 4];

#pragma unroll
    for (int mt = 0; mt < 2; ++mt) {
      v8f s0, s1;
#pragma unroll
      for (int r = 0; r < 8; ++r) { s0[r] = 0.f; s1[r] = 0.f; }
#pragma unroll
      for (int ks = 0; ks < 2; ++ks) {
        s0 = wmma_bf16(qf[mt][ks], kf[0][ks], s0);
        s1 = wmma_bf16(qf[mt][ks], kf[1][ks], s1);
      }
      // online softmax; a row lives across the 16 lanes of a half-wave
#pragma unroll
      for (int r = 0; r < 8; ++r) {
        float mx = fmaxf(s0[r], s1[r]);
        mx = fmaxf(mx, __shfl_xor(mx, 1, 32));
        mx = fmaxf(mx, __shfl_xor(mx, 2, 32));
        mx = fmaxf(mx, __shfl_xor(mx, 4, 32));
        mx = fmaxf(mx, __shfl_xor(mx, 8, 32));
        const float mn = fmaxf(mrun[mt][r], mx);
        const float sc = __expf(mrun[mt][r] - mn);
        const float e0 = __expf(s0[r] - mn);
        const float e1 = __expf(s1[r] - mn);
        float sm = e0 + e1;
        sm += __shfl_xor(sm, 1, 32);
        sm += __shfl_xor(sm, 2, 32);
        sm += __shfl_xor(sm, 4, 32);
        sm += __shfl_xor(sm, 8, 32);
        lrun[mt][r] = lrun[mt][r] * sc + sm;
        mrun[mt][r] = mn;
#pragma unroll
        for (int nt = 0; nt < 4; ++nt) O[mt][nt][r] *= sc;
        const int prow = mt * 16 + r + ((lane >> 4) << 3);
        Ps[wid][prow][lane & 15]        = f2bf(e0);
        Ps[wid][prow][16 + (lane & 15)] = f2bf(e1);
      }
    }
    __syncthreads();  // D-layout -> A-layout transpose via LDS
#pragma unroll
    for (int mt = 0; mt < 2; ++mt) {
      v16bf pf = load_a_frag(&Ps[wid][mt * 16][0], 32);
#pragma unroll
      for (int nt = 0; nt < 4; ++nt)
        O[mt][nt] = wmma_bf16(pf, vfr[nt], O[mt][nt]);
    }
  }

  // normalize and emit y[b,t,p, h*64+d] as bf16
#pragma unroll
  for (int mt = 0; mt < 2; ++mt)
#pragma unroll
    for (int nt = 0; nt < 4; ++nt) {
      const int dcol = nt * 16 + (lane & 15);
#pragma unroll
      for (int r = 0; r < 8; ++r) {
        const int row = qrow + mt * 16 + r + ((lane >> 4) << 3);
        Y[(size_t)(bt * P_TOK + row) * E_DIM + h * HD_DIM + dcol] =
            f2bf(O[mt][nt][r] / lrun[mt][r]);
      }
    }
}

// ---------------------------------------------------------------------------
extern "C" void kernel_launch(void* const* d_in, const int* in_sizes, int n_in,
                              void* d_out, int out_size, void* d_ws, size_t ws_size,
                              hipStream_t stream) {
  const float* x  = (const float*)d_in[0];
  const float* Wq = (const float*)d_in[1];
  const float* bq = (const float*)d_in[2];
  const float* Wk = (const float*)d_in[3];
  const float* bk = (const float*)d_in[4];
  const float* Wv = (const float*)d_in[5];
  const float* bv = (const float*)d_in[6];
  const float* Wo = (const float*)d_in[7];
  float* out = (float*)d_out;

  const size_t MN = (size_t)M_TOTAL * E_DIM;   // 25,165,824 elems
  __bf16* xb = (__bf16*)d_ws;                  // 5 bf16 planes ~ 252 MB total
  __bf16* qb = xb + MN;
  __bf16* kb = qb + MN;
  __bf16* vb = kb + MN;
  __bf16* yb = vb + MN;

  cvt_bf16_kernel<<<(int)(MN / 4 / 256), 256, 0, stream>>>(x, xb, (int)(MN / 4));

  dim3 ggrid(M_TOTAL / 128, E_DIM / 128);      // 256 x 6 blocks
  gemm_kernel<0><<<ggrid, 256, 0, stream>>>(xb, Wq, bq, qb, nullptr);
  gemm_kernel<1><<<ggrid, 256, 0, stream>>>(xb, Wk, bk, kb, nullptr);
  gemm_kernel<2><<<ggrid, 256, 0, stream>>>(xb, Wv, bv, vb, nullptr);

  attn_kernel<<<8 * 16 * HEADS, 256, 0, stream>>>(qb, kb, vb, yb);

  gemm_kernel<3><<<ggrid, 256, 0, stream>>>(yb, Wo, nullptr, nullptr, out);
}
